// Nonlocal_86517821211744
// MI455X (gfx1250) — matrix operand order repaired and checked by reading, not compiled, not run on hardware
//
#include <hip/hip_runtime.h>

// Non-local block (flash-attention formulation) for MI455X / gfx1250.
// N=4, C=512, CI=256, S=T*H*W=6272. All matmuls via v_wmma_f32_16x16x32_bf16.

#define N_  4
#define C_  512
#define CI_ 256
#define S_  6272
#define QT_PER_N 392        // S/16
#define EPS_ 1e-5f

typedef __attribute__((ext_vector_type(16))) __bf16 v16bf;
typedef __attribute__((ext_vector_type(8)))  float  v8f;

union V16 {
  v16bf v;
  uint4 q[2];
  __bf16 b[16];
  unsigned short s[16];
};

// Native f32 -> bf16 conversion (lowers to v_cvt_pk_bf16_f32 on gfx1250).
__device__ inline unsigned short f2bf(float f) {
  union { __bf16 b; unsigned short s; } u;
  u.b = (__bf16)f;
  return u.s;
}

// ---------------------------------------------------------------------------
// Kernel 1: QKV projections.  theta/phi -> [N,S,CI] bf16 (transposed), g -> [N,CI,S] bf16
// Each wave computes one 16x16 output tile with a K=512 WMMA loop.
// ---------------------------------------------------------------------------
__global__ __launch_bounds__(128) void proj_kernel(
    const float* __restrict__ x,
    const float* __restrict__ w_theta, const float* __restrict__ b_theta,
    const float* __restrict__ w_phi,   const float* __restrict__ b_phi,
    const float* __restrict__ w_g,     const float* __restrict__ b_g,
    unsigned short* __restrict__ theta_t,
    unsigned short* __restrict__ phi_t,
    unsigned short* __restrict__ g_buf)
{
  const int lane = threadIdx.x & 31;
  const int wv   = threadIdx.x >> 5;
  const int tile = blockIdx.x * 4 + wv;
  const int NT_S = N_ * QT_PER_N;                    // 1568 s-tiles
  const int proj = tile / (16 * NT_S);
  int rem = tile - proj * (16 * NT_S);
  const int ci0 = (rem / NT_S) * 16;
  rem = rem % NT_S;
  const int n  = rem / QT_PER_N;
  const int s0 = (rem % QT_PER_N) * 16;

  const float* W  = (proj == 0) ? w_theta : (proj == 1) ? w_phi : w_g;
  const float* Bi = (proj == 0) ? b_theta : (proj == 1) ? b_phi : b_g;

  const int m  = lane & 15;
  const int hi = lane >> 4;

  const float* xb = x + (size_t)n * C_ * S_;
  v8f acc = {};
  for (int kc = 0; kc < C_; kc += 32) {
    V16 a;                                            // A: W[ci0+m, kc..kc+31]
    const float* wrow = W + (size_t)(ci0 + m) * C_ + kc;
    #pragma unroll
    for (int h = 0; h < 8; ++h) a.b[h]     = (__bf16)wrow[hi * 8 + h];
    #pragma unroll
    for (int h = 0; h < 8; ++h) a.b[8 + h] = (__bf16)wrow[16 + hi * 8 + h];
    V16 b;                                            // B: x[kc+8*hi+(h&7)+16*(h>>3), s0+m]
    #pragma unroll
    for (int h = 0; h < 16; ++h)
      b.b[h] = (__bf16)xb[(size_t)(kc + 8 * hi + (h & 7) + 16 * (h >> 3)) * S_ + s0 + m];
    acc = __builtin_amdgcn_wmma_f32_16x16x32_bf16(false, a.v, false, b.v,
                                                  (short)0, acc, false, false);
  }
  const int s = s0 + m;
  if (proj == 2) {                                    // g: [N,CI,S]
    #pragma unroll
    for (int r = 0; r < 8; ++r) {
      const int ci = ci0 + r + 8 * hi;
      g_buf[((size_t)n * CI_ + ci) * S_ + s] = f2bf(acc[r] + Bi[ci]);
    }
  } else {                                            // theta/phi: [N,S,CI]
    unsigned short* d = (proj == 0) ? theta_t : phi_t;
    d += ((size_t)n * S_ + s) * CI_;
    #pragma unroll
    for (int r = 0; r < 8; ++r) {
      const int ci = ci0 + r + 8 * hi;
      d[ci] = f2bf(acc[r] + Bi[ci]);
    }
  }
}

// ---------------------------------------------------------------------------
// Kernel 2: flash attention.  Per wave: 16 queries, full CI=256 output.
// 32 keys/step: 2 score WMMAs x 8 K-chunks, online softmax (exp2 domain),
// probs -> LDS -> A-layout reload, 16 PV WMMAs.  out -> [N,S,CI] bf16.
// ---------------------------------------------------------------------------
__global__ __launch_bounds__(128) void flash_kernel(
    const unsigned short* __restrict__ theta_t,
    const unsigned short* __restrict__ phi_t,
    const unsigned short* __restrict__ g_buf,
    unsigned short* __restrict__ out_t)
{
  __shared__ unsigned short psm[4][16 * 32];          // per-wave 16x32 bf16 probs tile
  const int lane = threadIdx.x & 31;
  const int wv   = threadIdx.x >> 5;
  const int qt   = blockIdx.x * 4 + wv;
  const int n    = qt / QT_PER_N;
  const int t0   = (qt % QT_PER_N) * 16;
  const int m    = lane & 15;
  const int hi   = lane >> 4;

  const unsigned short* th = theta_t + (size_t)n * S_ * CI_;
  const unsigned short* ph = phi_t   + (size_t)n * S_ * CI_;
  const unsigned short* gb = g_buf   + (size_t)n * CI_ * S_;

  v8f acc[16];
  #pragma unroll
  for (int ct = 0; ct < 16; ++ct) acc[ct] = {};
  float mrow[8], lrow[8];
  #pragma unroll
  for (int r = 0; r < 8; ++r) { mrow[r] = -3.0e38f; lrow[r] = 0.f; }

  const float e = 1.44269504f * 0.0625f;              // log2(e) / sqrt(CI)

  for (int p0 = 0; p0 < S_; p0 += 32) {
    // prefetch next key block's phi rows while this block computes
    if (p0 + 32 < S_) {
      __builtin_prefetch(ph + (size_t)(p0 + 32 + m) * CI_, 0, 1);
      __builtin_prefetch(ph + (size_t)(p0 + 48 + m) * CI_, 0, 1);
      __builtin_prefetch(gb + (size_t)(m * 16) * S_ + p0 + 32, 0, 1);
    }
    v8f sa = {}, sb = {};
    #pragma unroll
    for (int kc = 0; kc < CI_; kc += 32) {
      V16 a;                                          // A: theta^T [16t x 32c]
      const unsigned short* arow = th + (size_t)(t0 + m) * CI_ + kc + hi * 16;
      a.q[0] = *(const uint4*)(arow);
      a.q[1] = *(const uint4*)(arow + 8);
      V16 b0, b1;                                     // B: phi [32c x 16p]
      const unsigned short* r0 = ph + (size_t)(p0 + m) * CI_ + kc + 16 * hi;
      b0.q[0] = *(const uint4*)(r0);
      b0.q[1] = *(const uint4*)(r0 + 8);
      const unsigned short* r1 = ph + (size_t)(p0 + 16 + m) * CI_ + kc + 16 * hi;
      b1.q[0] = *(const uint4*)(r1);
      b1.q[1] = *(const uint4*)(r1 + 8);
      sa = __builtin_amdgcn_wmma_f32_16x16x32_bf16(false, a.v, false, b0.v,
                                                   (short)0, sa, false, false);
      sb = __builtin_amdgcn_wmma_f32_16x16x32_bf16(false, a.v, false, b1.v,
                                                   (short)0, sb, false, false);
    }
    float alpha[8];
    #pragma unroll
    for (int r = 0; r < 8; ++r) {                     // online softmax per row
      float q0 = sa[r] * e, q1 = sb[r] * e;
      float cm = fmaxf(q0, q1);
      cm = fmaxf(cm, __shfl_xor(cm, 1));
      cm = fmaxf(cm, __shfl_xor(cm, 2));
      cm = fmaxf(cm, __shfl_xor(cm, 4));
      cm = fmaxf(cm, __shfl_xor(cm, 8));
      const float nm = fmaxf(mrow[r], cm);
      alpha[r] = exp2f(mrow[r] - nm);
      mrow[r]  = nm;
      const float p0v = exp2f(q0 - nm);
      const float p1v = exp2f(q1 - nm);
      float rs = p0v + p1v;
      rs += __shfl_xor(rs, 1);
      rs += __shfl_xor(rs, 2);
      rs += __shfl_xor(rs, 4);
      rs += __shfl_xor(rs, 8);
      lrow[r] = lrow[r] * alpha[r] + rs;
      psm[wv][(r + 8 * hi) * 32 + m]      = f2bf(p0v);   // row-major probs tile
      psm[wv][(r + 8 * hi) * 32 + 16 + m] = f2bf(p1v);
    }
    #pragma unroll
    for (int ct = 0; ct < 16; ++ct)
      #pragma unroll
      for (int r = 0; r < 8; ++r) acc[ct][r] *= alpha[r];

    V16 ap;                                           // reload probs in A layout
    const unsigned short* prow = &psm[wv][m * 32];
    ap.q[0] = *(const uint4*)(prow + hi * 16);
    ap.q[1] = *(const uint4*)(prow + hi * 16 + 8);

    #pragma unroll
    for (int ct = 0; ct < 16; ++ct) {                 // PV: D[t,c] += P x g^T
      V16 b;
      const unsigned short* grow = gb + (size_t)(ct * 16 + m) * S_ + p0 + 16 * hi;
      b.q[0] = *(const uint4*)(grow);
      b.q[1] = *(const uint4*)(grow + 8);
      acc[ct] = __builtin_amdgcn_wmma_f32_16x16x32_bf16(false, ap.v, false, b.v,
                                                        (short)0, acc[ct], false, false);
    }
  }
  #pragma unroll
  for (int r = 0; r < 8; ++r) lrow[r] = 1.0f / lrow[r];
  unsigned short* ob = out_t + (size_t)n * S_ * CI_;
  #pragma unroll
  for (int ct = 0; ct < 16; ++ct)
    #pragma unroll
    for (int r = 0; r < 8; ++r)
      ob[(size_t)(t0 + r + 8 * hi) * CI_ + ct * 16 + m] = f2bf(acc[ct][r] * lrow[r]);
}

// ---------------------------------------------------------------------------
// Kernel 3: output projection  p = w_out @ out + b_out   -> [N,C,S] fp32
// ---------------------------------------------------------------------------
__global__ __launch_bounds__(128) void outproj_kernel(
    const float* __restrict__ w_out, const float* __restrict__ b_out,
    const unsigned short* __restrict__ out_t,
    float* __restrict__ p)
{
  const int lane = threadIdx.x & 31;
  const int wv   = threadIdx.x >> 5;
  const int tile = blockIdx.x * 4 + wv;
  const int NT_S = N_ * QT_PER_N;
  const int c0 = (tile / NT_S) * 16;
  int rem = tile % NT_S;
  const int n  = rem / QT_PER_N;
  const int s0 = (rem % QT_PER_N) * 16;
  const int m  = lane & 15;
  const int hi = lane >> 4;

  const unsigned short* ob = out_t + (size_t)n * S_ * CI_;
  v8f acc = {};
  #pragma unroll
  for (int kc = 0; kc < CI_; kc += 32) {
    V16 a;
    const float* wrow = w_out + (size_t)(c0 + m) * CI_ + kc;
    #pragma unroll
    for (int h = 0; h < 8; ++h) a.b[h]     = (__bf16)wrow[hi * 8 + h];
    #pragma unroll
    for (int h = 0; h < 8; ++h) a.b[8 + h] = (__bf16)wrow[16 + hi * 8 + h];
    V16 b;
    const unsigned short* brow = ob + (size_t)(s0 + m) * CI_ + kc + 8 * hi;
    b.q[0] = *(const uint4*)(brow);
    b.q[1] = *(const uint4*)(brow + 16);
    acc = __builtin_amdgcn_wmma_f32_16x16x32_bf16(false, a.v, false, b.v,
                                                  (short)0, acc, false, false);
  }
  #pragma unroll
  for (int r = 0; r < 8; ++r) {
    const int c = c0 + r + 8 * hi;
    p[((size_t)n * C_ + c) * S_ + s0 + m] = acc[r] + b_out[c];
  }
}

// ---------------------------------------------------------------------------
// Kernel 4: deterministic per-channel BN statistics -> scale/shift
// ---------------------------------------------------------------------------
__global__ __launch_bounds__(256) void bn_reduce_kernel(
    const float* __restrict__ p, const float* __restrict__ gamma,
    const float* __restrict__ beta, float* __restrict__ stats)
{
  __shared__ float ssum[256];
  __shared__ float ssq[256];
  const int c = blockIdx.x;
  float s = 0.f, s2 = 0.f;
  for (int i = threadIdx.x; i < N_ * S_; i += 256) {
    const int n = i / S_, sp = i - n * S_;
    const float v = p[((size_t)n * C_ + c) * S_ + sp];
    s += v; s2 += v * v;
  }
  ssum[threadIdx.x] = s; ssq[threadIdx.x] = s2;
  __syncthreads();
  for (int off = 128; off > 0; off >>= 1) {
    if (threadIdx.x < off) {
      ssum[threadIdx.x] += ssum[threadIdx.x + off];
      ssq[threadIdx.x]  += ssq[threadIdx.x + off];
    }
    __syncthreads();
  }
  if (threadIdx.x == 0) {
    const float cnt  = (float)(N_ * S_);
    const float mean = ssum[0] / cnt;
    const float var  = ssq[0] / cnt - mean * mean;
    const float sc   = gamma[c] * rsqrtf(var + EPS_);
    stats[c]      = sc;
    stats[C_ + c] = beta[c] - mean * sc;
  }
}

// ---------------------------------------------------------------------------
// Kernel 5: y = x + p * scale[c] + shift[c]   (float4 vectorized)
// ---------------------------------------------------------------------------
__global__ __launch_bounds__(256) void bn_apply_kernel(
    const float* __restrict__ x, const float* __restrict__ p,
    const float* __restrict__ stats, float* __restrict__ y)
{
  const size_t i4 = (size_t)blockIdx.x * 256 + threadIdx.x;
  const size_t e  = i4 * 4;                           // S_ % 4 == 0 -> one channel per float4
  const int c = (int)((e / S_) % C_);
  const float4 xv = *(const float4*)(x + e);
  const float4 pv = *(const float4*)(p + e);
  const float sc = stats[c], sh = stats[C_ + c];
  float4 r;
  r.x = xv.x + pv.x * sc + sh;
  r.y = xv.y + pv.y * sc + sh;
  r.z = xv.z + pv.z * sc + sh;
  r.w = xv.w + pv.w * sc + sh;
  *(float4*)(y + e) = r;
}

// ---------------------------------------------------------------------------
extern "C" void kernel_launch(void* const* d_in, const int* in_sizes, int n_in,
                              void* d_out, int out_size, void* d_ws, size_t ws_size,
                              hipStream_t stream) {
  (void)in_sizes; (void)n_in; (void)out_size; (void)ws_size;
  const float* x       = (const float*)d_in[0];
  const float* w_theta = (const float*)d_in[1];
  const float* b_theta = (const float*)d_in[2];
  const float* w_phi   = (const float*)d_in[3];
  const float* b_phi   = (const float*)d_in[4];
  const float* w_g     = (const float*)d_in[5];
  const float* b_g     = (const float*)d_in[6];
  const float* w_out   = (const float*)d_in[7];
  const float* b_out   = (const float*)d_in[8];
  const float* gamma   = (const float*)d_in[9];
  const float* beta    = (const float*)d_in[10];
  float* y = (float*)d_out;

  char* ws = (char*)d_ws;
  const size_t szbf = (size_t)N_ * S_ * CI_ * sizeof(unsigned short);  // 12.8 MB
  unsigned short* theta_t = (unsigned short*)(ws);
  unsigned short* phi_t   = (unsigned short*)(ws + szbf);
  unsigned short* g_buf   = (unsigned short*)(ws + 2 * szbf);
  unsigned short* out_t   = (unsigned short*)(ws + 3 * szbf);
  float* p     = (float*)(ws + 4 * szbf);                              // 51.4 MB
  float* stats = (float*)(ws + 4 * szbf + (size_t)N_ * C_ * S_ * sizeof(float));

  // 3 projs * 16 ci-tiles * 1568 s-tiles = 75264 wave-tiles / 4 waves = 18816 blocks
  proj_kernel<<<18816, 128, 0, stream>>>(x, w_theta, b_theta, w_phi, b_phi,
                                         w_g, b_g, theta_t, phi_t, g_buf);
  // 1568 query tiles / 4 waves = 392 blocks
  flash_kernel<<<392, 128, 0, stream>>>(theta_t, phi_t, g_buf, out_t);
  // 32 c-tiles * 1568 s-tiles / 4 waves = 12544 blocks
  outproj_kernel<<<12544, 128, 0, stream>>>(w_out, b_out, out_t, p);
  bn_reduce_kernel<<<512, 256, 0, stream>>>(p, gamma, beta, stats);
  // 12,845,056 / 4 / 256 = 12544 blocks
  bn_apply_kernel<<<12544, 256, 0, stream>>>(x, p, stats, y);
}
